// ImprovedGNN_43121471652164
// MI455X (gfx1250) — matrix operand order repaired and checked by reading, not compiled, not run on hardware
//
#include <hip/hip_runtime.h>

typedef __bf16 bf16_t;
typedef __attribute__((ext_vector_type(16))) __bf16 v16bf;
typedef __attribute__((ext_vector_type(8)))  __bf16 v8bf;
typedef __attribute__((ext_vector_type(8)))  float  v8f;

#define NEG_SLOPE 0.2f
#define EPS 1e-16f

// ---------------------------------------------------------------- utilities
__global__ void fill_f32_kernel(float* __restrict__ p, float v, long long n) {
  long long t = (long long)blockIdx.x * blockDim.x + threadIdx.x;
  if (t < n) p[t] = v;
}
__global__ void fill_i32_kernel(int* __restrict__ p, int v, long long n) {
  long long t = (long long)blockIdx.x * blockDim.x + threadIdx.x;
  if (t < n) p[t] = v;
}
__global__ void cvt_f32_bf16_kernel(const float* __restrict__ s, bf16_t* __restrict__ d, long long n) {
  long long t = (long long)blockIdx.x * blockDim.x + threadIdx.x;
  if (t < n) d[t] = (bf16_t)s[t];
}
// Wt[n*K + k] = W[k*F + n]   (store B transposed so WMMA B-frag loads are contiguous)
__global__ void transpose_w_bf16_kernel(const float* __restrict__ W, bf16_t* __restrict__ Wt, int K, int F) {
  int t = blockIdx.x * blockDim.x + threadIdx.x;
  if (t < K * F) {
    int k = t % K, n = t / K;
    Wt[t] = (bf16_t)W[(size_t)k * F + n];
  }
}
// column sums of edge_attr [E, ED]
__global__ void ea_colsum_kernel(const float* __restrict__ ea, float* __restrict__ sum,
                                 long long E, int ED) {
  long long tid = (long long)blockIdx.x * blockDim.x + threadIdx.x;
  long long nthreads = (long long)gridDim.x * blockDim.x;
  int col = (int)(tid % ED);
  float acc = 0.f;
  for (long long r = tid / ED; r < E; r += nthreads / ED)
    acc += ea[r * ED + col];
  atomicAdd(&sum[col], acc);
}

// Ve[d,h] = sum_c We[d, h*C+c] * ae[h,c];  aeloop[h] = sum_d mean_ea[d] * Ve[d,h]
__global__ void compute_ve_kernel(const float* __restrict__ We, const float* __restrict__ ae,
                                  const float* __restrict__ easum, float einv,
                                  float* __restrict__ Ve, float* __restrict__ aeloop,
                                  int ED, int H, int C) {
  __shared__ float sVe[64];
  int t = threadIdx.x;
  if (t < ED * H) {
    int d = t / H, h = t % H;
    float s = 0.f;
    const float* wrow = We + (size_t)d * H * C + (size_t)h * C;
    const float* arow = ae + (size_t)h * C;
    for (int c = 0; c < C; ++c) s += wrow[c] * arow[c];
    sVe[t] = s;
    Ve[t] = s;
  }
  __syncthreads();
  if (t < H) {
    float s = 0.f;
    for (int d = 0; d < ED; ++d) s += easum[d] * einv * sVe[d * H + t];
    aeloop[t] = s;
  }
}

// ---------------------------------------------------------------- WMMA GEMM
// C[M,F] = A[M,K](bf16,row-major) * Bt[F,K](bf16, = B transposed), fp32 out.
// One wave per 16x16 tile; 8 waves/block cover 32(M) x 64(F).
__global__ __launch_bounds__(256)
void gemm_bf16_wmma_kernel(const bf16_t* __restrict__ A, const bf16_t* __restrict__ Bt,
                           float* __restrict__ Co, int M, int F, int K) {
  const int lane = threadIdx.x & 31;
  const int wave = threadIdx.x >> 5;
  const int g = lane >> 4;          // lane group 0/1
  const int r = lane & 15;
  const int row0 = blockIdx.y * 32 + (wave >> 2) * 16;
  const int col0 = blockIdx.x * 64 + (wave & 3) * 16;
  if (row0 >= M || col0 >= F) return;            // wave-uniform

  const int arow = (row0 + r) < M ? (row0 + r) : (M - 1);  // branchless clamp, EXEC stays full
  const bf16_t* Ap = A + (size_t)arow * K;
  const bf16_t* Bp = Bt + (size_t)(col0 + r) * K + 16 * g;

  v8f acc = {};
  for (int k0 = 0; k0 < K; k0 += 32) {
    // A frag (this code's contiguous k order): lane holds K = 16g..16g+15 (matches B frag)
    union { v16bf v; struct { v8bf lo; v8bf hi; } p; } a;
    a.p.lo = *(const v8bf*)(Ap + k0 + 16 * g);
    a.p.hi = *(const v8bf*)(Ap + k0 + 16 * g + 8);
    // B frag: lane holds column col0+r, K = 16g..16g+15 (contiguous in transposed W)
    v16bf b = *(const v16bf*)(Bp + k0);
    acc = __builtin_amdgcn_wmma_f32_16x16x32_bf16(false, a.v, false, b,
                                                  (short)0, acc, false, false);
  }
  // C/D layout: lanes 0-15 -> N=r, VGPR j -> M=j; lanes 16-31 -> M=j+8
  float* Cp = Co + (size_t)(row0 + 8 * g) * F + col0 + r;
#pragma unroll
  for (int j = 0; j < 8; ++j)
    if (row0 + 8 * g + j < M) Cp[(size_t)j * F] = acc[j];
}

// ---------------------------------------------------------------- attention
__global__ void node_att_kernel(const float* __restrict__ hprj,
                                const float* __restrict__ as_, const float* __restrict__ ad_,
                                float* __restrict__ a_s, float* __restrict__ a_d,
                                int Nn, int H, int C) {
  int t = blockIdx.x * blockDim.x + threadIdx.x;
  if (t >= Nn * H) return;
  int n = t / H, h = t % H;
  const float4* hp = (const float4*)(hprj + (size_t)n * H * C + (size_t)h * C);
  const float4* av = (const float4*)(as_ + (size_t)h * C);
  const float4* dv = (const float4*)(ad_ + (size_t)h * C);
  float ss = 0.f, sd = 0.f;
  for (int i = 0; i < C / 4; ++i) {
    float4 hv = hp[i], a = av[i], d = dv[i];
    ss += hv.x * a.x + hv.y * a.y + hv.z * a.z + hv.w * a.w;
    sd += hv.x * d.x + hv.y * d.y + hv.z * d.z + hv.w * d.w;
  }
  a_s[t] = ss; a_d[t] = sd;
}

__global__ void edge_ae_kernel(const float* __restrict__ ea, const float* __restrict__ Ve,
                               float* __restrict__ a_e, long long E, int ED, int H) {
  long long e = (long long)blockIdx.x * blockDim.x + threadIdx.x;
  if (e >= E) return;
  float acc[4] = {0.f, 0.f, 0.f, 0.f};
  const float* row = ea + e * ED;
  for (int d = 0; d < ED; ++d) {
    float v = row[d];
    for (int h = 0; h < H; ++h) acc[h] += v * Ve[d * H + h];
  }
  for (int h = 0; h < H; ++h) a_e[(size_t)e * H + h] = acc[h];
}

__device__ __forceinline__ int f32_key(float f) {
  int k = __float_as_int(f);
  return k >= 0 ? k : (k ^ 0x7fffffff);
}
__device__ __forceinline__ float key_f32(int k) {
  return __int_as_float(k >= 0 ? k : (k ^ 0x7fffffff));
}

__global__ void edge_pass1_kernel(const int* __restrict__ srcI, const int* __restrict__ dstI,
                                  const float* __restrict__ a_s, const float* __restrict__ a_d,
                                  const float* __restrict__ a_e, const float* __restrict__ aeloop,
                                  float* __restrict__ alpha, int* __restrict__ amax,
                                  long long E, int Nn, int H) {
  long long e = (long long)blockIdx.x * blockDim.x + threadIdx.x;
  if (e >= E + Nn) return;
  bool loop = e >= E;
  int s = loop ? (int)(e - E) : srcI[e];
  int d = loop ? (int)(e - E) : dstI[e];
  for (int h = 0; h < H; ++h) {
    float a = a_s[(size_t)s * H + h] + a_d[(size_t)d * H + h] +
              (loop ? aeloop[h] : a_e[(size_t)e * H + h]);
    a = a > 0.f ? a : NEG_SLOPE * a;          // leaky_relu
    alpha[(size_t)e * H + h] = a;
    atomicMax(&amax[(size_t)d * H + h], f32_key(a));
  }
}

__global__ void edge_pass2_kernel(const int* __restrict__ dstI, float* __restrict__ alpha,
                                  const int* __restrict__ amax, float* __restrict__ denom,
                                  long long E, int Nn, int H) {
  long long e = (long long)blockIdx.x * blockDim.x + threadIdx.x;
  if (e >= E + Nn) return;
  int d = (e >= E) ? (int)(e - E) : dstI[e];
  for (int h = 0; h < H; ++h) {
    float m = key_f32(amax[(size_t)d * H + h]);
    float ex = __expf(alpha[(size_t)e * H + h] - m);
    alpha[(size_t)e * H + h] = ex;
    atomicAdd(&denom[(size_t)d * H + h], ex);
  }
}

// one wave per edge; gather h[src] (L2-resident, 20MB), scatter-add into agg[dst]
__global__ __launch_bounds__(256)
void aggregate_kernel(const int* __restrict__ srcI, const int* __restrict__ dstI,
                      const float* __restrict__ hprj, const float* __restrict__ alpha,
                      const float* __restrict__ denom, float* __restrict__ agg,
                      long long E, int Nn, int H, int C) {
  const int lane = threadIdx.x & 31;
  const int wave = threadIdx.x >> 5;
  long long e = (long long)blockIdx.x * 8 + wave;
  if (e >= E + Nn) return;
  bool loop = e >= E;
  int s = loop ? (int)(e - E) : srcI[e];
  int d = loop ? (int)(e - E) : dstI[e];
  const int F = H * C;
  const float* hs = hprj + (size_t)s * F;
  float* ag = agg + (size_t)d * F;
  for (int f = lane; f < F; f += 32) {
    int h = f / C;
    float w = alpha[(size_t)e * H + h] / (denom[(size_t)d * H + h] + EPS);
    atomicAdd(&ag[f], hs[f] * w);
  }
}

__global__ void finalize_elu_kernel(const float* __restrict__ agg, const float* __restrict__ bias,
                                    bf16_t* __restrict__ hbf, long long n, int F) {
  long long t = (long long)blockIdx.x * blockDim.x + threadIdx.x;
  if (t >= n) return;
  float v = agg[t] + bias[(int)(t % F)];
  v = v > 0.f ? v : expm1f(v);                 // elu
  hbf[t] = (bf16_t)v;
}

__global__ void finalize_out_kernel(const float* __restrict__ agg, const float* __restrict__ bias,
                                    const float* __restrict__ lw, const float* __restrict__ lb,
                                    float* __restrict__ out, int Nn, int C) {
  int n = blockIdx.x * blockDim.x + threadIdx.x;
  if (n >= Nn) return;
  float s = 0.f;
  const float* row = agg + (size_t)n * C;
  for (int c = 0; c < C; ++c) s += (row[c] + bias[c]) * lw[c];
  out[n] = s + lb[0];
}

// ---------------------------------------------------------------- launcher
static inline int cdiv_ll(long long a, int b) { return (int)((a + b - 1) / b); }

extern "C" void kernel_launch(void* const* d_in, const int* in_sizes, int n_in,
                              void* d_out, int out_size, void* d_ws, size_t ws_size,
                              hipStream_t stream) {
  const float* x   = (const float*)d_in[0];
  const int*   ei  = (const int*)d_in[1];      // [2,E] int32
  const float* ea  = (const float*)d_in[2];
  const float* W[3]   = {(const float*)d_in[3],  (const float*)d_in[9],  (const float*)d_in[15]};
  const float* We[3]  = {(const float*)d_in[4],  (const float*)d_in[10], (const float*)d_in[16]};
  const float* as_[3] = {(const float*)d_in[5],  (const float*)d_in[11], (const float*)d_in[17]};
  const float* ad_[3] = {(const float*)d_in[6],  (const float*)d_in[12], (const float*)d_in[18]};
  const float* ae_[3] = {(const float*)d_in[7],  (const float*)d_in[13], (const float*)d_in[19]};
  const float* b_[3]  = {(const float*)d_in[8],  (const float*)d_in[14], (const float*)d_in[20]};
  const float* lin_w = (const float*)d_in[21];
  const float* lin_b = (const float*)d_in[22];

  const int Fin = 128, ED = 16, Cc = 64;
  const int N = in_sizes[0] / Fin;
  const long long E = in_sizes[1] / 2;
  const long long Et = E + N;

  // workspace carve-out (256B-aligned)
  char* ws = (char*)d_ws;
  size_t off = 0;
  auto take = [&](size_t bytes) -> char* {
    off = (off + 255) & ~(size_t)255;
    char* p = ws + off; off += bytes; return p;
  };
  float*  hprj  = (float*)take((size_t)N * 256 * 4);
  float*  agg   = (float*)take((size_t)N * 256 * 4);
  bf16_t* hbf   = (bf16_t*)take((size_t)N * 256 * 2);
  bf16_t* wbf   = (bf16_t*)take((size_t)256 * 256 * 2);
  float*  a_s   = (float*)take((size_t)N * 4 * 4);
  float*  a_d   = (float*)take((size_t)N * 4 * 4);
  int*    amax  = (int*)take((size_t)N * 4 * 4);
  float*  den   = (float*)take((size_t)N * 4 * 4);
  float*  a_e   = (float*)take((size_t)E * 4 * 4);
  float*  alph  = (float*)take((size_t)Et * 4 * 4);
  float*  easum = (float*)take(256);
  float*  Ve    = (float*)take(256);
  float*  aeloop= (float*)take(256);
  (void)ws_size; (void)n_in; (void)out_size;

  const int* srcI = ei;
  const int* dstI = ei + E;

  // edge_attr column sums (for self-loop fill 'mean')
  fill_f32_kernel<<<1, 64, 0, stream>>>(easum, 0.f, ED);
  ea_colsum_kernel<<<64, 256, 0, stream>>>(ea, easum, E, ED);

  // x -> bf16 activation buffer for layer-1 GEMM
  cvt_f32_bf16_kernel<<<cdiv_ll((long long)N * Fin, 256), 256, 0, stream>>>(x, hbf, (long long)N * Fin);

  const int Karr[3] = {Fin, 256, 256};
  const int Harr[3] = {4, 4, 1};

  for (int l = 0; l < 3; ++l) {
    const int K = Karr[l], H = Harr[l], F = H * Cc;

    // weights -> bf16, transposed [F,K]
    transpose_w_bf16_kernel<<<cdiv_ll((long long)K * F, 256), 256, 0, stream>>>(W[l], wbf, K, F);

    // hprj = hbf @ W   (WMMA)
    dim3 gg((F + 63) / 64, (N + 31) / 32);
    gemm_bf16_wmma_kernel<<<gg, 256, 0, stream>>>(hbf, wbf, hprj, N, F, K);

    // tiny folded edge projection: Ve = We . att_e ; loop-edge attention scalar
    compute_ve_kernel<<<1, 64, 0, stream>>>(We[l], ae_[l], easum, 1.f / (float)E, Ve, aeloop, ED, H, Cc);

    // per-node attention dots
    node_att_kernel<<<cdiv_ll((long long)N * H, 256), 256, 0, stream>>>(hprj, as_[l], ad_[l], a_s, a_d, N, H, Cc);

    // per-edge attention dots via Ve  (avoids materializing ea@We)
    edge_ae_kernel<<<cdiv_ll(E, 256), 256, 0, stream>>>(ea, Ve, a_e, E, ED, H);

    // reset segment state
    fill_i32_kernel<<<cdiv_ll((long long)N * H, 256), 256, 0, stream>>>(amax, (int)0x80000000, (long long)N * H);
    fill_f32_kernel<<<cdiv_ll((long long)N * H, 256), 256, 0, stream>>>(den, 0.f, (long long)N * H);
    fill_f32_kernel<<<cdiv_ll((long long)N * F, 256), 256, 0, stream>>>(agg, 0.f, (long long)N * F);

    // softmax over incoming edges
    edge_pass1_kernel<<<cdiv_ll(Et, 256), 256, 0, stream>>>(srcI, dstI, a_s, a_d, a_e, aeloop, alph, amax, E, N, H);
    edge_pass2_kernel<<<cdiv_ll(Et, 256), 256, 0, stream>>>(dstI, alph, amax, den, E, N, H);

    // message aggregation (one wave per edge)
    aggregate_kernel<<<cdiv_ll(Et, 8), 256, 0, stream>>>(srcI, dstI, hprj, alph, den, agg, E, N, H, Cc);

    if (l < 2) {
      finalize_elu_kernel<<<cdiv_ll((long long)N * F, 256), 256, 0, stream>>>(agg, b_[l], hbf, (long long)N * F, F);
    } else {
      finalize_out_kernel<<<cdiv_ll(N, 256), 256, 0, stream>>>(agg, b_[l], lin_w, lin_b, (float*)d_out, N, Cc);
    }
  }
}
